// RnnEncoder_14559939133889
// MI455X (gfx1250) — matrix operand order repaired and checked by reading, not compiled, not run on hardware
//
#include <hip/hip_runtime.h>
#include <hip/hip_bf16.h>
#include <math.h>

// ---------------------------------------------------------------------------
// RNN encoder for MI455X (gfx1250, wave32, WMMA).
//   h' = tanh(x_t W_ih^T + h W_hh^T + (b_ih+b_hh)), frozen past lengths[b].
//   FC head collapses (no activation between linears):
//       n = relu(h_new . v + c),  v = W2*W1 [H], c = b1.W2 + b2
//   outs[b] = #(t < len[b] : h_new . v + c > 0);   h_final = frozen h.
//
// Design: persistent 8-workgroup kernel, one WGP each; each WG owns a
// 64-column slice of H with its W_hh / W_ih slices resident in LDS
// (80 KB of the 320 KB WGP LDS) for all 2048 steps. Matmuls use
// v_wmma_f32_16x16x32_bf16 (fp32 accumulate); h kept fp32, double-buffered
// in global (L2-resident, 64 KB). Grid barrier per step via release/acquire
// atomics + s_sleep. Head dot reduced with half-wave shfl_xor + ds_add_f32.
// ---------------------------------------------------------------------------

typedef __bf16 v16bf __attribute__((ext_vector_type(16)));
typedef float  v8f   __attribute__((ext_vector_type(8)));

#define Bn   32
#define Tn   2048
#define Fn   128
#define Hn   512
#define FCn  128
#define NWG  8
#define NCOLS (Hn / NWG)        // 64 output columns per workgroup
#define WGT  256                // 8 wave32s per workgroup

// ---- workspace layout (bytes) ----------------------------------------------
// [0)          hbuf      : 2 * 32 * 512 f32   = 131072
// [131072)     partials  : 8 * 2048 * 32 f32  = 2097152
// [2228224)    bar       : 2 u32
// [2228480)    Whh bf16  : 512*512*2          = 524288
// [2752768)    Wih bf16  : 512*128*2          = 131072
// [2883840)    bias_c f32: 512*4
// [2885888)    vhead  f32: 512*4
// [2887936)    c      f32: 4
#define WS_HBUF   0
#define WS_PART   131072
#define WS_BAR    2228224
#define WS_WHH    2228480
#define WS_WIH    2752768
#define WS_BIAS   2883840
#define WS_V      2885888
#define WS_C      2887936
#define ZERO_F32  557064        // zero hbuf + partials + bar

// --- small prep kernels -----------------------------------------------------

__global__ void zero_f32_kernel(float* p, int n) {
    int i = blockIdx.x * blockDim.x + threadIdx.x;
    for (; i < n; i += gridDim.x * blockDim.x) p[i] = 0.0f;
}

__global__ void cvt_bf16_kernel(const float* __restrict__ src,
                                __bf16* __restrict__ dst, int n) {
    int i = blockIdx.x * blockDim.x + threadIdx.x;
    if (i < n) dst[i] = (__bf16)src[i];
}

// Collapse FC head + fold biases. One block of Hn threads.
__global__ void head_prep_kernel(const float* __restrict__ b_ih,
                                 const float* __restrict__ b_hh,
                                 const float* __restrict__ W1,   // [FC,H]
                                 const float* __restrict__ b1,   // [FC]
                                 const float* __restrict__ W2,   // [1,FC]
                                 const float* __restrict__ b2,   // [1]
                                 float* __restrict__ bias_c,
                                 float* __restrict__ vhead,
                                 float* __restrict__ cptr) {
    int h = threadIdx.x;                 // 0..511
    bias_c[h] = b_ih[h] + b_hh[h];
    float s = 0.0f;
    for (int f = 0; f < FCn; ++f) s += W2[f] * W1[f * Hn + h];
    vhead[h] = s;
    if (h == 0) {
        float c = b2[0];
        for (int f = 0; f < FCn; ++f) c += b1[f] * W2[f];
        *cptr = c;
    }
}

// --- WMMA fragment loaders (ISA 7.12.2 layouts, wave32) ---------------------
// A (16x32 bf16, MxK): lane covers K = kbase+half*8+[0..7] and +16.
__device__ __forceinline__ v16bf load_a_from_f32(const float* __restrict__ rowp,
                                                 int kbase, int half) {
    v16bf a;
    const float* p0 = rowp + kbase + half * 8;
    const float* p1 = p0 + 16;
#pragma unroll
    for (int i = 0; i < 8; ++i) {
        a[i]     = (__bf16)p0[i];
        a[8 + i] = (__bf16)p1[i];
    }
    return a;
}
// B (32x16 bf16, KxN) from row-major [N x K] weight slice (i.e. W^T GEMM):
// lane covers K = kbase + half*8 + [0..7] and +16.
__device__ __forceinline__ v16bf load_b_from_lds(const __bf16* __restrict__ rowp,
                                                 int kbase, int half) {
    v16bf b;
    const __bf16* p = rowp + kbase + half * 8;
#pragma unroll
    for (int i = 0; i < 8; ++i) { b[i] = p[i]; b[8 + i] = p[16 + i]; }
    return b;
}

// --- grid-wide sense-reversing barrier --------------------------------------
__device__ __forceinline__ void grid_sync(unsigned* bar, int nwg) {
    __syncthreads();
    if (threadIdx.x == 0) {
        __threadfence();  // make this WG's h stores device-visible
        unsigned gen = __hip_atomic_load(&bar[1], __ATOMIC_RELAXED,
                                         __HIP_MEMORY_SCOPE_AGENT);
        unsigned arrived = __hip_atomic_fetch_add(&bar[0], 1u, __ATOMIC_ACQ_REL,
                                                  __HIP_MEMORY_SCOPE_AGENT);
        if (arrived == (unsigned)(nwg - 1)) {
            __hip_atomic_store(&bar[0], 0u, __ATOMIC_RELAXED,
                               __HIP_MEMORY_SCOPE_AGENT);
            __hip_atomic_fetch_add(&bar[1], 1u, __ATOMIC_RELEASE,
                                   __HIP_MEMORY_SCOPE_AGENT);
        } else {
            while (__hip_atomic_load(&bar[1], __ATOMIC_ACQUIRE,
                                     __HIP_MEMORY_SCOPE_AGENT) == gen)
                __builtin_amdgcn_s_sleep(1);
        }
    }
    __syncthreads();
}

// --- persistent recurrent kernel --------------------------------------------
__global__ __launch_bounds__(WGT)
void rnn_persistent_kernel(const float* __restrict__ x,        // [B,T,F]
                           const int*   __restrict__ lengths,  // [B]
                           const __bf16* __restrict__ Whh,     // [H,H] (n,k)
                           const __bf16* __restrict__ Wih,     // [H,F] (n,k)
                           const float* __restrict__ bias_c,   // [H]
                           const float* __restrict__ vhead,    // [H]
                           float* __restrict__ hbuf,           // [2][B,H]
                           float* __restrict__ partials,       // [NWG][T,B]
                           unsigned* __restrict__ bar,
                           float* __restrict__ h_final_out) {  // [B,H]
    __shared__ __bf16 sWhh[NCOLS * Hn];   // 64 KB, resident all 2048 steps
    __shared__ __bf16 sWih[NCOLS * Fn];   // 16 KB
    __shared__ float  sdot[Bn];

    const int wg   = blockIdx.x;
    const int tid  = threadIdx.x;
    const int n0g  = wg * NCOLS;

    // Stage this WG's weight slices into LDS (128-bit chunks).
    {
        const uint4* g = (const uint4*)(Whh + (size_t)n0g * Hn);
        uint4* s = (uint4*)sWhh;
        for (int i = tid; i < NCOLS * Hn / 8; i += WGT) s[i] = g[i];
        const uint4* g2 = (const uint4*)(Wih + (size_t)n0g * Fn);
        uint4* s2 = (uint4*)sWih;
        for (int i = tid; i < NCOLS * Fn / 8; i += WGT) s2[i] = g2[i];
    }
    __syncthreads();

    // One 16x16 C tile per wave: 2 M-tiles x 4 N-tiles.
    const int wave  = tid >> 5;
    const int lane  = tid & 31;
    const int half  = lane >> 4;       // 0: rows r, 1: rows r+8
    const int l16   = lane & 15;
    const int mt    = wave >> 2;       // 0..1  (batch rows 0-15 / 16-31)
    const int nt    = wave & 3;        // 0..3  (16-col groups in slice)
    const int m     = mt * 16 + l16;   // A-fragment row (batch index)
    const int nloc  = nt * 16 + l16;   // B-fragment row (local out column)
    const int nglob = n0g + nloc;      // this lane's output column n

    const float bias = bias_c[nglob];
    const float vn   = vhead[nglob];
    int len_r[8];
#pragma unroll
    for (int r = 0; r < 8; ++r) len_r[r] = lengths[mt * 16 + r + 8 * half];

    for (int t = 0; t < Tn; ++t) {
        const float* hin  = hbuf + (size_t)(t & 1) * (Bn * Hn);
        float*       hout = hbuf + (size_t)((t + 1) & 1) * (Bn * Hn);
        if (tid < Bn) sdot[tid] = 0.0f;
        __syncthreads();

        v8f acc = {};
        // x_t @ W_ih^T   (K = 128)
        const float* xrow = x + ((size_t)m * Tn + t) * Fn;
        __builtin_prefetch(xrow + Fn, 0, 1);   // next step's x row
#pragma unroll
        for (int kt = 0; kt < Fn / 32; ++kt) {
            v16bf a = load_a_from_f32(xrow, kt * 32, half);
            v16bf b = load_b_from_lds(&sWih[nloc * Fn], kt * 32, half);
            acc = __builtin_amdgcn_wmma_f32_16x16x32_bf16(
                false, a, false, b, (short)0, acc, false, false);
        }
        // h @ W_hh^T     (K = 512)
        const float* hrow = hin + (size_t)m * Hn;
#pragma unroll
        for (int kt = 0; kt < Hn / 32; ++kt) {
            v16bf a = load_a_from_f32(hrow, kt * 32, half);
            v16bf b = load_b_from_lds(&sWhh[nloc * Hn], kt * 32, half);
            acc = __builtin_amdgcn_wmma_f32_16x16x32_bf16(
                false, a, false, b, (short)0, acc, false, false);
        }

        // tanh, head-dot partials, masked h update.
        float hv[8];
#pragma unroll
        for (int r = 0; r < 8; ++r) hv[r] = tanhf(acc[r] + bias);

#pragma unroll
        for (int r = 0; r < 8; ++r) {
            float s = hv[r] * vn;          // reduce over 16 cols (half-wave)
            s += __shfl_xor(s, 1, 16);
            s += __shfl_xor(s, 2, 16);
            s += __shfl_xor(s, 4, 16);
            s += __shfl_xor(s, 8, 16);
            if (l16 == 0) atomicAdd(&sdot[mt * 16 + r + 8 * half], s);
        }
#pragma unroll
        for (int r = 0; r < 8; ++r) {
            int row = mt * 16 + r + 8 * half;
            size_t idx = (size_t)row * Hn + nglob;
            hout[idx] = (t < len_r[r]) ? hv[r] : hin[idx];   // freeze past len
        }

        __syncthreads();
        if (tid < Bn)
            partials[((size_t)wg * Tn + t) * Bn + tid] = sdot[tid];
        grid_sync(bar, NWG);
    }

    // T is even -> final frozen h lives in buffer 0. Each WG owns its slice.
#pragma unroll
    for (int r = 0; r < 8; ++r) {
        int row = mt * 16 + r + 8 * half;
        size_t idx = (size_t)row * Hn + nglob;
        h_final_out[idx] = hbuf[idx];
    }
}

// --- finalize: count positives per batch ------------------------------------
__global__ void finalize_kernel(const float* __restrict__ partials,
                                const int* __restrict__ lengths,
                                const float* __restrict__ cptr,
                                float* __restrict__ outs) {
    const int b = blockIdx.x;            // 32 blocks
    const int len = lengths[b];
    const float c = *cptr;
    int cnt = 0;
    for (int t = threadIdx.x; t < len; t += blockDim.x) {
        float s = c;
#pragma unroll
        for (int w = 0; w < NWG; ++w)
            s += partials[((size_t)w * Tn + t) * Bn + b];
        cnt += (s > 0.0f) ? 1 : 0;
    }
    __shared__ int red[256];
    red[threadIdx.x] = cnt;
    __syncthreads();
    for (int s = 128; s > 0; s >>= 1) {
        if (threadIdx.x < s) red[threadIdx.x] += red[threadIdx.x + s];
        __syncthreads();
    }
    if (threadIdx.x == 0) outs[b] = (float)red[0];
}

// --- launch -----------------------------------------------------------------
extern "C" void kernel_launch(void* const* d_in, const int* in_sizes, int n_in,
                              void* d_out, int out_size, void* d_ws, size_t ws_size,
                              hipStream_t stream) {
    const float* x    = (const float*)d_in[0];   // [B,T,F]
    const int* lens   = (const int*)  d_in[1];   // [B]
    const float* W_ih = (const float*)d_in[2];   // [H,F]
    const float* W_hh = (const float*)d_in[3];   // [H,H]
    const float* b_ih = (const float*)d_in[4];
    const float* b_hh = (const float*)d_in[5];
    const float* W1   = (const float*)d_in[6];   // [FC,H]
    const float* b1   = (const float*)d_in[7];
    const float* W2   = (const float*)d_in[8];   // [1,FC]
    const float* b2   = (const float*)d_in[9];

    float* out = (float*)d_out;                  // [0..31]=counts, [32..)=h_final
    char* ws = (char*)d_ws;
    float*    hbuf     = (float*)   (ws + WS_HBUF);
    float*    partials = (float*)   (ws + WS_PART);
    unsigned* bar      = (unsigned*)(ws + WS_BAR);
    __bf16*   Whh_bf   = (__bf16*)  (ws + WS_WHH);
    __bf16*   Wih_bf   = (__bf16*)  (ws + WS_WIH);
    float*    bias_c   = (float*)   (ws + WS_BIAS);
    float*    vhead    = (float*)   (ws + WS_V);
    float*    cptr     = (float*)   (ws + WS_C);

    zero_f32_kernel<<<512, 256, 0, stream>>>((float*)ws, ZERO_F32);
    cvt_bf16_kernel<<<(Hn * Hn + 255) / 256, 256, 0, stream>>>(W_hh, Whh_bf, Hn * Hn);
    cvt_bf16_kernel<<<(Hn * Fn + 255) / 256, 256, 0, stream>>>(W_ih, Wih_bf, Hn * Fn);
    head_prep_kernel<<<1, Hn, 0, stream>>>(b_ih, b_hh, W1, b1, W2, b2,
                                           bias_c, vhead, cptr);
    rnn_persistent_kernel<<<NWG, WGT, 0, stream>>>(x, lens, Whh_bf, Wih_bf,
                                                   bias_c, vhead, hbuf, partials,
                                                   bar, out + Bn);
    finalize_kernel<<<Bn, 256, 0, stream>>>(partials, lens, cptr, out);
}
